// GAT_14998025797698
// MI455X (gfx1250) — matrix operand order repaired and checked by reading, not compiled, not run on hardware
//
#include <hip/hip_runtime.h>
#include <hip/hip_bf16.h>
#include <stdint.h>

#define NNODES   100000
#define NPADROWS 100032   // NNODES rounded up to 64
#define NODE_D   64
#define NH       4
#define NC       64
#define HC       256      // NH*NC
#define NLAYERS  3
#define LN_EPS   1e-5f

typedef __attribute__((ext_vector_type(16))) __bf16 v16bf;
typedef __attribute__((ext_vector_type(8)))  float  v8f;

__device__ __forceinline__ float lrelu(float x) { return x > 0.f ? x : 0.2f * x; }

__device__ __forceinline__ unsigned short f32_to_bf16_rne(float f) {
    unsigned u = __float_as_uint(f);
    unsigned r = u + 0x7FFFu + ((u >> 16) & 1u);
    return (unsigned short)(r >> 16);
}

__device__ __forceinline__ void atomic_add_f32(float* p, float v) {
    unsafeAtomicAdd(p, v);   // native global_atomic_add_f32
}

// ---------------- f32 -> bf16 convert (with zero pad) ----------------
__global__ void k_f32_to_bf16(const float* __restrict__ src,
                              unsigned short* __restrict__ dst, int n, int npad) {
    int i = blockIdx.x * blockDim.x + threadIdx.x;
    if (i < npad) dst[i] = (i < n) ? f32_to_bf16_rne(src[i]) : (unsigned short)0;
}

// ---------------- GEMM: Y[NPADROWS,256] = Xbf[NPADROWS,64] @ Wbf[64,256] ----------------
// Block: 256 threads = 8 waves; block tile 64(M) x 32(N); wave tile 16x16; K=64 => 2 WMMAs.
// Fully branch-free: inputs zero-padded, outputs padded, stores unconditional.
__global__ __launch_bounds__(256) void k_gemm_bf16(
    const unsigned short* __restrict__ X,   // [NPADROWS, 64] bf16 (zero padded)
    const unsigned short* __restrict__ W,   // [64, 256] bf16
    float* __restrict__ Y) {                // [NPADROWS, 256]
    __shared__ __align__(16) unsigned short Wt[32 * 64];   // transposed: Wt[n][k]

    const int tid  = threadIdx.x;
    const int nblk = blockIdx.y;            // 0..7 -> 32 columns each

    // Cooperative transpose of W column tile into LDS (coalesced along n)
    {
        const int n = tid & 31;
        const int kb = tid >> 5;            // 0..7
        #pragma unroll
        for (int it = 0; it < 8; ++it) {
            int k = kb + it * 8;
            Wt[n * 64 + k] = W[k * 256 + nblk * 32 + n];
        }
    }
    __syncthreads();

    const int wave = tid >> 5;              // 0..7
    const int lane = tid & 31;
    const int wm = wave >> 1;               // 0..3
    const int wn = wave & 1;                // 0..1
    const int m0  = blockIdx.x * 64 + wm * 16;
    const int n0g = nblk * 32 + wn * 16;
    const int t    = lane & 15;
    const int half = lane >> 4;

    union Frag { v16bf v; uint4 q[2]; } a0, a1, b0, b1;

    // A 16x32 bf16: lanes 0-15 hold K {0..7,16..23}, lanes 16-31 hold K {8..15,24..31}
    const uint4* xrow = (const uint4*)(X + (size_t)(m0 + t) * 64);  // 8 b128 chunks
    a0.q[0] = xrow[0 + half];
    a0.q[1] = xrow[2 + half];
    a1.q[0] = xrow[4 + half];
    a1.q[1] = xrow[6 + half];

    // B 32x16 bf16: lanes 0-15 hold K {0..7,16..23}, lanes 16-31 hold K {8..15,24..31} (col = lane&15)
    const uint4* brow = (const uint4*)(&Wt[(wn * 16 + t) * 64]);
    b0.q[0] = brow[0 + half];
    b0.q[1] = brow[2 + half];
    b1.q[0] = brow[4 + half];
    b1.q[1] = brow[6 + half];

    v8f c = {0.f, 0.f, 0.f, 0.f, 0.f, 0.f, 0.f, 0.f};
    c = __builtin_amdgcn_wmma_f32_16x16x32_bf16(false, a0.v, false, b0.v, (short)0, c, false, false);
    c = __builtin_amdgcn_wmma_f32_16x16x32_bf16(false, a1.v, false, b1.v, (short)0, c, false, false);

    union { v8f v; float f[8]; } cu; cu.v = c;
    float* yp = Y + (size_t)(m0 + 8 * half) * 256 + n0g + t;   // rows m0+8h .. +7
    #pragma unroll
    for (int r = 0; r < 8; ++r) yp[(size_t)r * 256] = cu.f[r];
}

// ---------------- Edge pass 1: logits + segment max ----------------
// One wave per edge; lane owns 8 channels; head = lane>>3.
__global__ __launch_bounds__(256) void k_edge_logits(
    const int* __restrict__ ei, int E, int Etot,
    const float* __restrict__ xl, const float* __restrict__ xr,
    const float* __restrict__ att,
    float* __restrict__ logits, unsigned* __restrict__ mmax) {
    int w = blockIdx.x * 8 + (threadIdx.x >> 5);
    if (w >= Etot) return;
    const int lane = threadIdx.x & 31;
    int s, d;
    if (w < E) { s = ei[w]; d = ei[E + w]; } else { s = d = w - E; }

    const size_t bs = (size_t)s * HC + lane * 8;
    const size_t bd = (size_t)d * HC + lane * 8;
    float4 a0 = *(const float4*)(xl + bs);
    float4 a1 = *(const float4*)(xl + bs + 4);
    float4 r0 = *(const float4*)(xr + bd);
    float4 r1 = *(const float4*)(xr + bd + 4);
    const float4* at = (const float4*)(att + lane * 8);
    float4 t0 = at[0], t1 = at[1];

    float p = 0.f;
    p += t0.x * lrelu(a0.x + r0.x);
    p += t0.y * lrelu(a0.y + r0.y);
    p += t0.z * lrelu(a0.z + r0.z);
    p += t0.w * lrelu(a0.w + r0.w);
    p += t1.x * lrelu(a1.x + r1.x);
    p += t1.y * lrelu(a1.y + r1.y);
    p += t1.z * lrelu(a1.z + r1.z);
    p += t1.w * lrelu(a1.w + r1.w);

    p += __shfl_xor(p, 1);
    p += __shfl_xor(p, 2);
    p += __shfl_xor(p, 4);

    if ((lane & 7) == 0) {
        int h = lane >> 3;
        logits[(size_t)w * NH + h] = p;
        unsigned bits = __float_as_uint(p);
        unsigned enc = (bits & 0x80000000u) ? ~bits : (bits | 0x80000000u); // monotonic map
        atomicMax(&mmax[(size_t)d * NH + h], enc);
    }
}

// ---------------- Edge pass 2: ex = exp(l-m); scatter ex and ex*xl[src] ----------------
__global__ __launch_bounds__(256) void k_edge_aggregate(
    const int* __restrict__ ei, int E, int Etot,
    const float* __restrict__ xl, const float* __restrict__ logits,
    const unsigned* __restrict__ mmax,
    float* __restrict__ denom, float* __restrict__ acc) {
    int w = blockIdx.x * 8 + (threadIdx.x >> 5);
    if (w >= Etot) return;
    const int lane = threadIdx.x & 31;
    int s, d;
    if (w < E) { s = ei[w]; d = ei[E + w]; } else { s = d = w - E; }

    const int h = lane >> 3;
    float lg = logits[(size_t)w * NH + h];
    unsigned um = mmax[(size_t)d * NH + h];
    unsigned mb = (um & 0x80000000u) ? (um & 0x7FFFFFFFu) : ~um;
    float m = __uint_as_float(mb);
    float ex = __expf(lg - m);

    if ((lane & 7) == 0) atomic_add_f32(&denom[(size_t)d * NH + h], ex);

    const size_t bs = (size_t)s * HC + lane * 8;
    float4 a0 = *(const float4*)(xl + bs);
    float4 a1 = *(const float4*)(xl + bs + 4);
    float* ap = acc + (size_t)d * HC + lane * 8;
    atomic_add_f32(ap + 0, ex * a0.x);
    atomic_add_f32(ap + 1, ex * a0.y);
    atomic_add_f32(ap + 2, ex * a0.z);
    atomic_add_f32(ap + 3, ex * a0.w);
    atomic_add_f32(ap + 4, ex * a1.x);
    atomic_add_f32(ap + 5, ex * a1.y);
    atomic_add_f32(ap + 6, ex * a1.z);
    atomic_add_f32(ap + 7, ex * a1.w);
}

// ---------------- Per node: head-mean + bias; accumulate LN stats ----------------
__global__ __launch_bounds__(256) void k_combine(
    const float* __restrict__ acc, const float* __restrict__ denom,
    const float* __restrict__ bias, float* __restrict__ xcur,
    float* __restrict__ lnstats) {
    int idx = blockIdx.x * 256 + threadIdx.x;
    float v = 0.f;
    if (idx < NNODES * NC) {
        int n = idx >> 6, c = idx & 63;
        float sum = 0.f;
        #pragma unroll
        for (int h = 0; h < NH; ++h)
            sum += acc[(size_t)n * HC + h * NC + c] / (denom[(size_t)n * NH + h] + 1e-16f);
        v = 0.25f * sum + bias[c];
        xcur[idx] = v;
    }
    float s1 = v, s2 = v * v;
    #pragma unroll
    for (int o = 1; o < 32; o <<= 1) { s1 += __shfl_xor(s1, o); s2 += __shfl_xor(s2, o); }
    __shared__ float sh1[8], sh2[8];
    int lane = threadIdx.x & 31, wv = threadIdx.x >> 5;
    if (lane == 0) { sh1[wv] = s1; sh2[wv] = s2; }
    __syncthreads();
    if (threadIdx.x == 0) {
        float a = 0.f, b = 0.f;
        #pragma unroll
        for (int i = 0; i < 8; ++i) { a += sh1[i]; b += sh2[i]; }
        atomic_add_f32(&lnstats[0], a);
        atomic_add_f32(&lnstats[1], b);
    }
}

// ---------------- Graph LayerNorm apply + fused bf16 conversion of result ----------------
__global__ __launch_bounds__(256) void k_layernorm(
    const float* __restrict__ xcur, const float* __restrict__ lnstats,
    const float* __restrict__ w, const float* __restrict__ b,
    float* __restrict__ out, unsigned short* __restrict__ out_bf) {
    int idx = blockIdx.x * blockDim.x + threadIdx.x;
    if (idx >= NNODES * NC) return;
    const float inv = 1.0f / (float)(NNODES * NC);
    float mu  = lnstats[0] * inv;
    float var = lnstats[1] * inv - mu * mu;
    float rs  = rsqrtf(var + LN_EPS);
    int c = idx & 63;
    float v = (xcur[idx] - mu) * rs * w[c] + b[c];
    out[idx] = v;
    out_bf[idx] = f32_to_bf16_rne(v);   // feeds next layer's WMMA GEMM
}

extern "C" void kernel_launch(void* const* d_in, const int* in_sizes, int n_in,
                              void* d_out, int out_size, void* d_ws, size_t ws_size,
                              hipStream_t stream) {
    const float* x0   = (const float*)d_in[0];
    const int*   ei   = (const int*)d_in[1];
    const float* Wl   = (const float*)d_in[2];
    const float* Wr   = (const float*)d_in[3];
    const float* att  = (const float*)d_in[4];
    const float* bias = (const float*)d_in[5];
    const float* lnw  = (const float*)d_in[6];
    const float* lnb  = (const float*)d_in[7];
    float* out = (float*)d_out;

    const int E    = in_sizes[1] / 2;
    const int Etot = E + NNODES;

    char* wp = (char*)d_ws;
    auto alloc = [&](size_t bytes) -> char* {
        char* p = wp; wp += (bytes + 255) & ~(size_t)255; return p;
    };
    unsigned short* xbf  = (unsigned short*)alloc((size_t)NPADROWS * NODE_D * 2);
    unsigned short* wbf  = (unsigned short*)alloc((size_t)NLAYERS * 2 * 64 * 256 * 2); // all Wl|Wr
    float* xl     = (float*)alloc((size_t)NPADROWS * HC * 4);
    float* xr     = (float*)alloc((size_t)NPADROWS * HC * 4);   // re-used as acc
    float* logits = (float*)alloc((size_t)Etot * NH * 4);
    // mmax, denom, lnstats contiguous -> one memset
    unsigned* mmax = (unsigned*)alloc((size_t)NNODES * NH * 4);
    float* denom  = (float*)alloc((size_t)NNODES * NH * 4);
    float* lnstats = (float*)alloc(256);
    float* xcur   = (float*)alloc((size_t)NNODES * NODE_D * 4);
    const size_t zero_span = (char*)lnstats + 256 - (char*)mmax;

    const int nx  = NNODES * NODE_D;
    const int nxp = NPADROWS * NODE_D;
    const int eb  = (Etot + 7) / 8;               // 8 waves (edges) per block
    const int WELT = NLAYERS * 2 * 64 * 256;      // all weight elements

    // One-time conversions this call: x (zero-padded) and all 6 weight matrices
    k_f32_to_bf16<<<(nxp + 255) / 256, 256, 0, stream>>>(x0, xbf, nx, nxp);
    k_f32_to_bf16<<<(WELT + 255) / 256, 256, 0, stream>>>(Wl, wbf, 64 * 256 * NLAYERS,
                                                          64 * 256 * NLAYERS);
    k_f32_to_bf16<<<(WELT + 255) / 256, 256, 0, stream>>>(
        Wr, wbf + (size_t)NLAYERS * 64 * 256, 64 * 256 * NLAYERS, 64 * 256 * NLAYERS);

    for (int layer = 0; layer < NLAYERS; ++layer) {
        const unsigned short* wlb = wbf + (size_t)layer * 64 * 256;
        const unsigned short* wrb = wbf + (size_t)(NLAYERS + layer) * 64 * 256;

        // xl = x@Wl, xr = x@Wr via WMMA (branch-free, padded)
        dim3 gg(NPADROWS / 64, 8);
        k_gemm_bf16<<<gg, 256, 0, stream>>>(xbf, wlb, xl);
        k_gemm_bf16<<<gg, 256, 0, stream>>>(xbf, wrb, xr);

        // zero mmax + denom + lnstats in one shot (mmax==0 is the ordered-uint floor)
        hipMemsetAsync(mmax, 0, zero_span, stream);
        k_edge_logits<<<eb, 256, 0, stream>>>(ei, E, Etot, xl, xr,
                                              att + (size_t)layer * HC, logits, mmax);

        // acc aliases xr (done being read)
        hipMemsetAsync(xr, 0, (size_t)NNODES * HC * 4, stream);
        k_edge_aggregate<<<eb, 256, 0, stream>>>(ei, E, Etot, xl, logits, mmax, denom, xr);

        k_combine<<<(NNODES * NC + 255) / 256, 256, 0, stream>>>(
            xr, denom, bias + (size_t)layer * NC, xcur, lnstats);

        float* dst = (layer == NLAYERS - 1) ? out : xcur;
        k_layernorm<<<(NNODES * NC + 255) / 256, 256, 0, stream>>>(
            xcur, lnstats, lnw + (size_t)layer * NC, lnb + (size_t)layer * NC, dst, xbf);
    }
}
